// EfficientTransformerEncoder_62594853372030
// MI455X (gfx1250) — matrix operand order repaired and checked by reading, not compiled, not run on hardware
//
#include <hip/hip_runtime.h>
#include <hip/hip_bf16.h>

// ---------------------------------------------------------------------------
// Types for CDNA5 WMMA (wave32, 16x16x32 bf16 -> f32)
// ---------------------------------------------------------------------------
typedef __attribute__((ext_vector_type(16))) __bf16 v16bf;
typedef __attribute__((ext_vector_type(8)))  __bf16 bf16x8;
typedef __attribute__((ext_vector_type(8)))  float  v8f;

union U16 { v16bf v; bf16x8 h[2]; };

#define D_MODEL 256
#define NHEADS  8
#define DH      32
#define NLAYERS 3
#define DFF     1024
#define WIN     64
#define SEQ     1024
#define BATCH   32
#define NTOK    (BATCH * SEQ)   // 32768

// ---------------------------------------------------------------------------
// CDNA5 async global->LDS copy (ASYNCcnt-tracked), with sync fallback
// ---------------------------------------------------------------------------
#if __has_builtin(__builtin_amdgcn_global_load_async_to_lds_b128) && \
    __has_builtin(__builtin_amdgcn_s_wait_asynccnt)
#define ASYNC_LDS 1
// builtin expects: (int4(vector_size 16) AS1*, int4(vector_size 16) AS3*, imm, imm)
typedef int v4i_gcc __attribute__((vector_size(16)));
typedef __attribute__((address_space(1))) v4i_gcc gv4i_t;
typedef __attribute__((address_space(3))) v4i_gcc lv4i_t;
#endif

__device__ __forceinline__ void copy16_tile(const __bf16* g, __bf16* l) {
#ifdef ASYNC_LDS
    // generic shared ptr low 32 bits == LDS byte offset == AS3 representation;
    // generic global ptr == AS1 representation.
    __builtin_amdgcn_global_load_async_to_lds_b128(
        (gv4i_t*)(unsigned long long)g,
        (lv4i_t*)(unsigned int)(unsigned long long)l,
        0, 0);
#else
    *(bf16x8*)l = *(const bf16x8*)g;
#endif
}

__device__ __forceinline__ void copy_wait_all() {
#ifdef ASYNC_LDS
    __builtin_amdgcn_s_wait_asynccnt(0);
#endif
}

__device__ __forceinline__ float warp_red_add16(float v) {
    for (int o = 1; o < 16; o <<= 1) v += __shfl_xor(v, o, 32);
    return v;
}
__device__ __forceinline__ float warp_red_max16(float v) {
    for (int o = 1; o < 16; o <<= 1) v = fmaxf(v, __shfl_xor(v, o, 32));
    return v;
}
__device__ __forceinline__ float warp_red_add32(float v) {
    for (int o = 1; o < 32; o <<= 1) v += __shfl_xor(v, o, 32);
    return v;
}

// ---------------------------------------------------------------------------
// fp32 -> bf16 weight conversion
// ---------------------------------------------------------------------------
__global__ __launch_bounds__(256) void cvt_bf16_kernel(const float* __restrict__ src,
                                                       __bf16* __restrict__ dst, int n) {
    for (int i = blockIdx.x * blockDim.x + threadIdx.x; i < n; i += gridDim.x * blockDim.x)
        dst[i] = (__bf16)src[i];
}

// ---------------------------------------------------------------------------
// Prep: features [B, D, L] -> x [B*L, D] (+ sinusoidal PE), fp32 + bf16
// ---------------------------------------------------------------------------
__global__ __launch_bounds__(256) void prep_kernel(const float* __restrict__ features,
                                                   float* __restrict__ xf,
                                                   __bf16* __restrict__ xb) {
    const int t = blockIdx.x;              // token 0..32767
    const int b = t >> 10;
    const int l = t & 1023;
    const int d = threadIdx.x;             // channel 0..255
    float f = features[((long)(b * D_MODEL + d)) * SEQ + l];
    float freq = expf(-(float)(d & ~1) * (9.210340371976184f / (float)D_MODEL));
    float ang  = (float)l * freq;
    float pe   = (d & 1) ? cosf(ang) : sinf(ang);
    float v = f + pe;
    long idx = (long)t * D_MODEL + d;
    xf[idx] = v;
    xb[idx] = (__bf16)v;
}

// ---------------------------------------------------------------------------
// Tiled bf16 WMMA GEMM:  C[M,N] = A[M,K] * W[N,K]^T + bias[N]
// Block tile 128(M) x 64(N), 8 waves, each wave 32x32 via 2x2 WMMA tiles.
// Double-buffered LDS with async global->LDS copies overlapping WMMA.
// gelu != 0 applies exact GELU. Either output pointer may be null.
// ---------------------------------------------------------------------------
__global__ __launch_bounds__(256) void gemm_bf16_kernel(
    const __bf16* __restrict__ A, int lda,
    const __bf16* __restrict__ W, int ldw,
    const float* __restrict__ bias,
    float* __restrict__ Cf, __bf16* __restrict__ Cb,
    int N, int gelu)
{
    __shared__ __align__(16) __bf16 As[2][128][40];   // +8 pad halves per row
    __shared__ __align__(16) __bf16 Bs[2][64][40];

    const int tid  = threadIdx.x;
    const int lane = tid & 31;
    const int wv   = tid >> 5;
    const int wm   = wv & 3;         // 4 waves along M
    const int wn   = wv >> 2;        // 2 waves along N
    const int hi   = lane >> 4;
    const int lr   = lane & 15;
    const long mb  = (long)blockIdx.x * 128;
    const int  nb  = blockIdx.y * 64;
    const int  K   = lda;

    v8f acc[2][2] = {};

    const int arow = tid >> 1, acol = (tid & 1) * 16;   // 128 rows x 32 halves
    const int brow = tid >> 2, bcol = (tid & 3) * 8;    // 64 rows  x 32 halves

    const __bf16* Ar = A + (mb + arow) * (long)lda;
    const __bf16* Wr = W + (long)(nb + brow) * ldw;

    __builtin_prefetch(Wr, 0, 1);   // global_prefetch of the weight tile

    // prologue: stage K-step 0 into buffer 0
    copy16_tile(Ar + acol,     &As[0][arow][acol]);
    copy16_tile(Ar + acol + 8, &As[0][arow][acol + 8]);
    copy16_tile(Wr + bcol,     &Bs[0][brow][bcol]);

    int ib = 0;
    for (int k0 = 0; k0 < K; k0 += 32, ib ^= 1) {
        copy_wait_all();
        __syncthreads();            // buffer ib fully populated; prior compute done

        if (k0 + 32 < K) {          // stage next step into the other buffer
            copy16_tile(Ar + k0 + 32 + acol,     &As[ib ^ 1][arow][acol]);
            copy16_tile(Ar + k0 + 32 + acol + 8, &As[ib ^ 1][arow][acol + 8]);
            copy16_tile(Wr + k0 + 32 + bcol,     &Bs[ib ^ 1][brow][bcol]);
        }

        // A fragment: lanes 0-15 -> K 0..15; lanes 16-31 -> K 16..31
        v16bf af[2], bfr[2];
        for (int mt = 0; mt < 2; ++mt) {
            int r = wm * 32 + mt * 16 + lr;
            U16 u;
            u.h[0] = *(const bf16x8*)&As[ib][r][16 * hi];
            u.h[1] = *(const bf16x8*)&As[ib][r][16 * hi + 8];
            af[mt] = u.v;
        }
        // B fragment: lanes 0-15 -> K 0..15; lanes 16-31 -> K 16..31
        for (int nt = 0; nt < 2; ++nt) {
            int r = wn * 32 + nt * 16 + lr;
            U16 u;
            u.h[0] = *(const bf16x8*)&Bs[ib][r][16 * hi];
            u.h[1] = *(const bf16x8*)&Bs[ib][r][16 * hi + 8];
            bfr[nt] = u.v;
        }
        for (int mt = 0; mt < 2; ++mt)
            for (int nt = 0; nt < 2; ++nt)
                acc[mt][nt] = __builtin_amdgcn_wmma_f32_16x16x32_bf16(
                    false, af[mt], false, bfr[nt], (short)0, acc[mt][nt], false, false);
    }

    for (int mt = 0; mt < 2; ++mt)
        for (int nt = 0; nt < 2; ++nt)
            for (int r = 0; r < 8; ++r) {
                long grow = mb + wm * 32 + mt * 16 + r + 8 * hi;
                int  gcol = nb + wn * 32 + nt * 16 + lr;
                float v = acc[mt][nt][r] + bias[gcol];
                if (gelu) v = 0.5f * v * (1.0f + erff(v * 0.70710678118654752f));
                if (Cf) Cf[grow * (long)N + gcol] = v;
                if (Cb) Cb[grow * (long)N + gcol] = (__bf16)v;
            }
}

// ---------------------------------------------------------------------------
// Windowed attention. One block per 64-token window, wave w == head w.
// qkv bf16 [NTOK, 768] laid out q|k|v, channel = head*32 + d.
// Writes attn output bf16 [NTOK, 256].
// ---------------------------------------------------------------------------
__global__ __launch_bounds__(256) void attn_kernel(const __bf16* __restrict__ qkv,
                                                   __bf16* __restrict__ attnb) {
    __shared__ __align__(16) __bf16 vT[NHEADS][DH][72];   // v transposed per head
    __shared__ __align__(16) __bf16 Ps[NHEADS][16][WIN];  // softmax strip per head

    const int tid  = threadIdx.x;
    const int h    = tid >> 5;       // head
    const int lane = tid & 31;
    const int hi   = lane >> 4;
    const int lr   = lane & 15;
    const long tb  = (long)blockIdx.x * WIN;

    // stage v^T: v[t][d] -> vT[d][t]
    for (int t = lane; t < WIN; t += 32) {
        const __bf16* vrow = qkv + (tb + t) * 768 + 512 + h * DH;
        bf16x8 v0 = *(const bf16x8*)(vrow);
        bf16x8 v1 = *(const bf16x8*)(vrow + 8);
        bf16x8 v2 = *(const bf16x8*)(vrow + 16);
        bf16x8 v3 = *(const bf16x8*)(vrow + 24);
        for (int j = 0; j < 8; ++j) {
            vT[h][j][t]      = v0[j];
            vT[h][j + 8][t]  = v1[j];
            vT[h][j + 16][t] = v2[j];
            vT[h][j + 24][t] = v3[j];
        }
    }
    __syncthreads();

    // k fragments (B operand of q*k^T): lane -> token n = nt*16+lr, K=d
    v16bf kf[4];
    for (int nt = 0; nt < 4; ++nt) {
        const __bf16* kp = qkv + (tb + nt * 16 + lr) * 768 + 256 + h * DH + hi * 16;
        U16 u;
        u.h[0] = *(const bf16x8*)kp;
        u.h[1] = *(const bf16x8*)(kp + 8);
        kf[nt] = u.v;
    }

    const float scale = 0.17677669529663687f;  // 1/sqrt(32)

    for (int mi = 0; mi < 4; ++mi) {
        // q fragment (A operand), row = token tb + mi*16 + lr
        const __bf16* qp = qkv + (tb + mi * 16 + lr) * 768 + h * DH;
        U16 uq;
        uq.h[0] = *(const bf16x8*)(qp + 16 * hi);
        uq.h[1] = *(const bf16x8*)(qp + 16 * hi + 8);
        v16bf qf = uq.v;

        v8f s[4];
        for (int nt = 0; nt < 4; ++nt) {
            v8f z = {};
            s[nt] = __builtin_amdgcn_wmma_f32_16x16x32_bf16(
                false, qf, false, kf[nt], (short)0, z, false, false);
        }
        for (int nt = 0; nt < 4; ++nt)
            for (int r = 0; r < 8; ++r) s[nt][r] *= scale;

        // softmax per row (row m = r + 8*hi; 64 cols spread over 16 lanes x 4 tiles)
        for (int r = 0; r < 8; ++r) {
            float m = fmaxf(fmaxf(s[0][r], s[1][r]), fmaxf(s[2][r], s[3][r]));
            m = warp_red_max16(m);
            float e0 = expf(s[0][r] - m), e1 = expf(s[1][r] - m);
            float e2 = expf(s[2][r] - m), e3 = expf(s[3][r] - m);
            float sum = warp_red_add16(e0 + e1 + e2 + e3);
            float inv = 1.0f / sum;
            int row = r + 8 * hi;
            Ps[h][row][lr]      = (__bf16)(e0 * inv);
            Ps[h][row][lr + 16] = (__bf16)(e1 * inv);
            Ps[h][row][lr + 32] = (__bf16)(e2 * inv);
            Ps[h][row][lr + 48] = (__bf16)(e3 * inv);
        }
        __syncthreads();

        // O strip [16 x 32] = P [16 x 64] * v [64 x 32]
        v8f o0 = {}, o1 = {};
        for (int kt = 0; kt < 2; ++kt) {
            const __bf16* pp = &Ps[h][lr][kt * 32 + 16 * hi];
            U16 up;
            up.h[0] = *(const bf16x8*)pp;
            up.h[1] = *(const bf16x8*)(pp + 8);
            v16bf pf = up.v;
            for (int nt2 = 0; nt2 < 2; ++nt2) {
                const __bf16* vp = &vT[h][nt2 * 16 + lr][kt * 32 + 16 * hi];
                U16 uv;
                uv.h[0] = *(const bf16x8*)vp;
                uv.h[1] = *(const bf16x8*)(vp + 8);
                if (nt2 == 0)
                    o0 = __builtin_amdgcn_wmma_f32_16x16x32_bf16(
                        false, pf, false, uv.v, (short)0, o0, false, false);
                else
                    o1 = __builtin_amdgcn_wmma_f32_16x16x32_bf16(
                        false, pf, false, uv.v, (short)0, o1, false, false);
            }
        }
        for (int r = 0; r < 8; ++r) {
            long tok = tb + mi * 16 + r + 8 * hi;
            attnb[tok * D_MODEL + h * DH + lr]      = (__bf16)o0[r];
            attnb[tok * D_MODEL + h * DH + 16 + lr] = (__bf16)o1[r];
        }
        __syncthreads();
    }
}

// ---------------------------------------------------------------------------
// Fused residual + LayerNorm, one token per wave (8 tokens/block).
// res may be null (pure LN). Writes fp32 (in place OK) and bf16 copies.
// ---------------------------------------------------------------------------
__global__ __launch_bounds__(256) void ln_kernel(const float* __restrict__ x,
                                                 const float* __restrict__ res,
                                                 const float* __restrict__ sc,
                                                 const float* __restrict__ bi,
                                                 float* __restrict__ xo,
                                                 __bf16* __restrict__ xob) {
    const int wv = threadIdx.x >> 5, lane = threadIdx.x & 31;
    const long t = (long)blockIdx.x * 8 + wv;
    float v[8];
    float s = 0.f;
    for (int j = 0; j < 8; ++j) {
        int ch = lane + 32 * j;
        float a = x[t * D_MODEL + ch];
        if (res) a += res[t * D_MODEL + ch];
        v[j] = a;
        s += a;
    }
    s = warp_red_add32(s);
    float mean = s * (1.0f / D_MODEL);
    float q = 0.f;
    for (int j = 0; j < 8; ++j) { float d = v[j] - mean; q += d * d; }
    q = warp_red_add32(q);
    float inv = rsqrtf(q * (1.0f / D_MODEL) + 1e-5f);
    for (int j = 0; j < 8; ++j) {
        int ch = lane + 32 * j;
        float o = (v[j] - mean) * inv * sc[ch] + bi[ch];
        if (xo)  xo[t * D_MODEL + ch]  = o;
        if (xob) xob[t * D_MODEL + ch] = (__bf16)o;
    }
}

// ---------------------------------------------------------------------------
// Final LN + transpose to output [B, C, H*W]
// ---------------------------------------------------------------------------
__global__ __launch_bounds__(256) void final_kernel(const float* __restrict__ x,
                                                    const float* __restrict__ sc,
                                                    const float* __restrict__ bi,
                                                    float* __restrict__ out) {
    const int wv = threadIdx.x >> 5, lane = threadIdx.x & 31;
    const long t = (long)blockIdx.x * 8 + wv;
    const int b = (int)(t >> 10);
    const int l = (int)(t & 1023);
    float v[8];
    float s = 0.f;
    for (int j = 0; j < 8; ++j) {
        int ch = lane + 32 * j;
        float a = x[t * D_MODEL + ch];
        v[j] = a;
        s += a;
    }
    s = warp_red_add32(s);
    float mean = s * (1.0f / D_MODEL);
    float q = 0.f;
    for (int j = 0; j < 8; ++j) { float d = v[j] - mean; q += d * d; }
    q = warp_red_add32(q);
    float inv = rsqrtf(q * (1.0f / D_MODEL) + 1e-5f);
    for (int j = 0; j < 8; ++j) {
        int ch = lane + 32 * j;
        float o = (v[j] - mean) * inv * sc[ch] + bi[ch];
        out[((long)(b * D_MODEL + ch)) * SEQ + l] = o;
    }
}

// ---------------------------------------------------------------------------
// Host launcher
// ---------------------------------------------------------------------------
extern "C" void kernel_launch(void* const* d_in, const int* in_sizes, int n_in,
                              void* d_out, int out_size, void* d_ws, size_t ws_size,
                              hipStream_t stream) {
    (void)in_sizes; (void)n_in; (void)out_size; (void)ws_size;

    const float* features   = (const float*)d_in[0];
    const float* in_proj_w  = (const float*)d_in[1];
    const float* in_proj_b  = (const float*)d_in[2];
    const float* out_proj_w = (const float*)d_in[3];
    const float* out_proj_b = (const float*)d_in[4];
    const float* w1         = (const float*)d_in[5];
    const float* b1         = (const float*)d_in[6];
    const float* w2         = (const float*)d_in[7];
    const float* b2         = (const float*)d_in[8];
    const float* ln1_s      = (const float*)d_in[9];
    const float* ln1_b      = (const float*)d_in[10];
    const float* ln2_s      = (const float*)d_in[11];
    const float* ln2_b      = (const float*)d_in[12];
    const float* fnorm_s    = (const float*)d_in[13];
    const float* fnorm_b    = (const float*)d_in[14];
    float* out = (float*)d_out;

    // workspace layout
    char* base = (char*)d_ws;
    size_t off = 0;
    auto alloc = [&](size_t bytes) -> char* {
        off = (off + 255) & ~(size_t)255;
        char* p = base + off;
        off += bytes;
        return p;
    };
    float*  xf    = (float*) alloc((size_t)NTOK * D_MODEL * 4);
    __bf16* xb    = (__bf16*)alloc((size_t)NTOK * D_MODEL * 2);
    __bf16* qkvb  = (__bf16*)alloc((size_t)NTOK * 3 * D_MODEL * 2);
    __bf16* attnb = (__bf16*)alloc((size_t)NTOK * D_MODEL * 2);
    float*  yf    = (float*) alloc((size_t)NTOK * D_MODEL * 4);
    __bf16* hb    = (__bf16*)alloc((size_t)NTOK * DFF * 2);
    __bf16* wib   = (__bf16*)alloc((size_t)NLAYERS * 3 * D_MODEL * D_MODEL * 2);
    __bf16* wob   = (__bf16*)alloc((size_t)NLAYERS * D_MODEL * D_MODEL * 2);
    __bf16* w1b   = (__bf16*)alloc((size_t)NLAYERS * DFF * D_MODEL * 2);
    __bf16* w2b   = (__bf16*)alloc((size_t)NLAYERS * D_MODEL * DFF * 2);

    // weight conversion
    {
        int n_wi = NLAYERS * 3 * D_MODEL * D_MODEL;
        int n_wo = NLAYERS * D_MODEL * D_MODEL;
        int n_w1 = NLAYERS * DFF * D_MODEL;
        int n_w2 = NLAYERS * D_MODEL * DFF;
        cvt_bf16_kernel<<<(n_wi + 2047) / 2048, 256, 0, stream>>>(in_proj_w, wib, n_wi);
        cvt_bf16_kernel<<<(n_wo + 2047) / 2048, 256, 0, stream>>>(out_proj_w, wob, n_wo);
        cvt_bf16_kernel<<<(n_w1 + 2047) / 2048, 256, 0, stream>>>(w1, w1b, n_w1);
        cvt_bf16_kernel<<<(n_w2 + 2047) / 2048, 256, 0, stream>>>(w2, w2b, n_w2);
    }

    // prep: features + PE
    prep_kernel<<<NTOK, 256, 0, stream>>>(features, xf, xb);

    for (int i = 0; i < NLAYERS; ++i) {
        // QKV projection -> bf16
        gemm_bf16_kernel<<<dim3(NTOK / 128, (3 * D_MODEL) / 64), 256, 0, stream>>>(
            xb, D_MODEL, wib + (size_t)i * 3 * D_MODEL * D_MODEL, D_MODEL,
            in_proj_b + (size_t)i * 3 * D_MODEL, nullptr, qkvb, 3 * D_MODEL, 0);

        // windowed attention
        attn_kernel<<<NTOK / WIN, 256, 0, stream>>>(qkvb, attnb);

        // out projection -> f32
        gemm_bf16_kernel<<<dim3(NTOK / 128, D_MODEL / 64), 256, 0, stream>>>(
            attnb, D_MODEL, wob + (size_t)i * D_MODEL * D_MODEL, D_MODEL,
            out_proj_b + (size_t)i * D_MODEL, yf, nullptr, D_MODEL, 0);

        // x = LN1(x + att)
        ln_kernel<<<NTOK / 8, 256, 0, stream>>>(
            xf, yf, ln1_s + (size_t)i * D_MODEL, ln1_b + (size_t)i * D_MODEL, xf, xb);

        // FFN up + GELU -> bf16
        gemm_bf16_kernel<<<dim3(NTOK / 128, DFF / 64), 256, 0, stream>>>(
            xb, D_MODEL, w1b + (size_t)i * DFF * D_MODEL, D_MODEL,
            b1 + (size_t)i * DFF, nullptr, hb, DFF, 1);

        // FFN down -> f32
        gemm_bf16_kernel<<<dim3(NTOK / 128, D_MODEL / 64), 256, 0, stream>>>(
            hb, DFF, w2b + (size_t)i * D_MODEL * DFF, DFF,
            b2 + (size_t)i * D_MODEL, yf, nullptr, D_MODEL, 0);

        // x = LN2(x + h)
        ln_kernel<<<NTOK / 8, 256, 0, stream>>>(
            xf, yf, ln2_s + (size_t)i * D_MODEL, ln2_b + (size_t)i * D_MODEL, xf, xb);
    }

    // final LN + transpose back to [B, C, H, W]
    final_kernel<<<NTOK / 8, 256, 0, stream>>>(xf, fnorm_s, fnorm_b, out);
}
